// Model_39178691674157
// MI455X (gfx1250) — matrix operand order repaired and checked by reading, not compiled, not run on hardware
//
#include <hip/hip_runtime.h>
#include <hip/hip_bf16.h>
#include <stdint.h>

// ---------------------------------------------------------------------------
// GRU(HID=1024) x T steps + projection to 96 cols, tuned for MI455X (gfx1250).
//  * init kernel: pack Wh/Wd to bf16 pairs in *fragment-major* order (each
//    lane's 8 WMMA B-operand dwords contiguous) in workspace.
//  * persistent GRU kernel (32 WGs): stage 197KB Wh slice into LDS with
//    global_load_async_to_lds_b128 (ASYNCcnt), then per step do the 1024x96
//    matvec with v_wmma_f32_16x16x32_bf16 (split-K over 8 waves, LDS reduce),
//    VALU gates, 2KB h exchange through L2 + grid barrier.
//  * projection: dense WMMA bf16 GEMM, B tile staged in LDS via ds_load_b128.
// ---------------------------------------------------------------------------

#define HID     1024
#define IN_DIMC 8
#define G3      96          // 3 gates x 32 units per WG
#define HB      32          // hidden units per WG
#define NWG     32          // HID / HB
#define KP      512         // HID/2 packed bf16 pair-rows
#define FRAGW   (KP * G3)   // 49152 u32 words per WG slice / per Wd image

typedef __attribute__((ext_vector_type(16))) __bf16 v16bf;
typedef __attribute__((ext_vector_type(8)))  float  v8f;

union Frag { v16bf v; unsigned u[8]; };

__device__ __forceinline__ unsigned short f2bf(float f) {
  union { float f; unsigned u; } x; x.f = f;
  unsigned u = x.u;
  unsigned lsb = (u >> 16) & 1u;
  return (unsigned short)((u + 0x7FFFu + lsb) >> 16);   // round-to-nearest-even
}
__device__ __forceinline__ unsigned pack2(float a, float b) {
  return (unsigned)f2bf(a) | ((unsigned)f2bf(b) << 16);
}

// fragment-major index: (kp = K pair-row, cl = column 0..95) ->
//   [kchunk][nt][lane = half*16 + n][j]  with 8 contiguous u32 per lane.
__device__ __forceinline__ int frag_index(int kp, int cl) {
  int kchunk = kp >> 4, jp = kp & 15;
  int half = jp >> 3, jj = jp & 7;
  int nt = cl >> 4, n = cl & 15;
  return ((((kchunk * 6 + nt) << 5) + (half << 4) + n) << 3) + jj;
}

// ---------------------------------------------------------------------------
// init: zero counter + h buffers; build fragment-major bf16 images of Wd and
// of each workgroup's Wh slice.
// ---------------------------------------------------------------------------
__global__ void init_ws_kernel(const float* __restrict__ Wh,   // [1024,3072]
                               const float* __restrict__ Wd,   // [1024,96]
                               unsigned* __restrict__ counter,
                               unsigned* __restrict__ hbuf,    // [2][512]
                               unsigned* __restrict__ Wdp,     // [FRAGW]
                               unsigned* __restrict__ WhpG) {  // [NWG][FRAGW]
  int tid = blockIdx.x * blockDim.x + threadIdx.x;
  int nthr = gridDim.x * blockDim.x;
  if (tid == 0) *counter = 0u;
  if (tid < 2 * KP) hbuf[tid] = 0u;

  for (int i = tid; i < KP * G3; i += nthr) {
    int kp = i / G3, c = i % G3;
    Wdp[frag_index(kp, c)] = pack2(Wd[(2 * kp) * G3 + c], Wd[(2 * kp + 1) * G3 + c]);
  }
  for (int i = tid; i < KP * 3 * HID; i += nthr) {
    int kp = i / (3 * HID), gc = i % (3 * HID);
    int gate = gc >> 10, unit = gc & 1023;
    int wg = unit >> 5, cl = (gate << 5) + (unit & 31);
    unsigned v = pack2(Wh[(2 * kp) * (3 * HID) + gc],
                       Wh[(2 * kp + 1) * (3 * HID) + gc]);
    WhpG[wg * FRAGW + frag_index(kp, cl)] = v;
  }
}

// ---------------------------------------------------------------------------
// Persistent GRU recurrence.
// ---------------------------------------------------------------------------
__global__ void __launch_bounds__(256, 1)
gru_kernel(const float* __restrict__ x,      // [T,8]
           const float* __restrict__ Wx,     // [8,3072]
           const float* __restrict__ bx,     // [3072]
           const float* __restrict__ bh,     // [3072]
           const unsigned* __restrict__ WhpG,// [NWG][FRAGW] frag-major bf16
           unsigned* __restrict__ counter,
           unsigned* __restrict__ hbuf,      // [2][512] packed bf16 pairs
           unsigned short* __restrict__ hs,  // [T,1024] bf16 bits
           int T) {
  extern __shared__ unsigned smem[];
  unsigned*        WhsF   = smem;                          // FRAGW u32
  unsigned*        hp     = WhsF + FRAGW;                  // 512 u32
  float*           part   = (float*)(hp + KP);             // 8*96
  float*           gx_s   = part + 8 * G3;                 // 96
  float*           gh_s   = gx_s + G3;                     // 96
  float*           Wxs    = gh_s + G3;                     // 8*96
  float*           bxs    = Wxs + 8 * G3;                  // 96
  float*           bhs    = bxs + G3;                      // 96
  float*           xs     = bhs + G3;                      // 8
  float*           h_loc  = xs + IN_DIMC;                  // 32
  unsigned short*  hbits  = (unsigned short*)(h_loc + HB); // 32

  const int tid  = threadIdx.x;
  const int wg   = blockIdx.x;          // 0..31
  const int u0   = wg * HB;
  const int lane = tid & 31;
  const int wave = tid >> 5;            // 0..7

  // ---- async DMA the pre-packed Wh slice into LDS (ASYNCcnt path) ----
  {
    const unsigned long long gbase =
        (unsigned long long)(const void*)(WhpG + (size_t)wg * FRAGW);
    const unsigned lds0 = (unsigned)(uintptr_t)(void*)WhsF;
#pragma unroll 4
    for (int r = 0; r < FRAGW / 4 / 256; ++r) {   // 48 x b128 per thread
      unsigned goff = (unsigned)(tid + r * 256) * 16u;
      unsigned loff = lds0 + goff;
      asm volatile("global_load_async_to_lds_b128 %0, %1, %2"
                   :: "v"(loff), "v"(goff), "s"(gbase) : "memory");
    }
    asm volatile("s_wait_asynccnt 0x0" ::: "memory");
  }
  // Wx slice + biases (tiny)
  for (int i = tid; i < IN_DIMC * G3; i += 256) {
    int d = i / G3, cl = i % G3;
    int gc = (cl >> 5) * HID + u0 + (cl & 31);
    Wxs[i] = Wx[d * (3 * HID) + gc];
  }
  if (tid < G3) {
    int gc = (tid >> 5) * HID + u0 + (tid & 31);
    bxs[tid] = bx[gc];
    bhs[tid] = bh[gc];
  }
  if (tid < HB) h_loc[tid] = 0.0f;
  __syncthreads();

  const int  half   = lane >> 4;            // K-half select
  const bool a_lane = (lane & 15) == 0;     // only row M=0 of A is live
  const int  n      = lane & 15;            // B/D column within tile

  for (int t = 0; t < T; ++t) {
    const unsigned* hcur = hbuf + (t & 1) * KP;
    unsigned*       hnxt = hbuf + ((t + 1) & 1) * KP;

    for (int i = tid; i < KP; i += 256) hp[i] = hcur[i];
    if (tid < IN_DIMC) xs[tid] = (t == 0) ? 0.0f : x[(t - 1) * IN_DIMC + tid];
    __syncthreads();

    // ---- gh = h @ WhSlice via WMMA; wave w covers K in [w*128, w*128+128) ----
    v8f acc[6] = {};
#pragma unroll
    for (int kc = 0; kc < 4; ++kc) {
      const int kchunk = wave * 4 + kc;        // K chunk of 32 (16 pair rows)
      const int kp0 = kchunk * 16;
      Frag A;
#pragma unroll
      for (int j = 0; j < 8; ++j) A.u[j] = 0u;
      if (a_lane) {                            // lanes 0 and 16 carry row 0
        const uint4 a0 = *(const uint4*)(hp + kp0 + half * 8);
        const uint4 a1 = *(const uint4*)(hp + kp0 + half * 8 + 4);
        A.u[0] = a0.x; A.u[1] = a0.y; A.u[2] = a0.z; A.u[3] = a0.w;
        A.u[4] = a1.x; A.u[5] = a1.y; A.u[6] = a1.z; A.u[7] = a1.w;
      }
#pragma unroll
      for (int nt = 0; nt < 6; ++nt) {
        const unsigned* fb = WhsF + (kchunk * 6 + nt) * 256 + lane * 8;
        const uint4 b0 = *(const uint4*)(fb);
        const uint4 b1 = *(const uint4*)(fb + 4);
        Frag B;
        B.u[0] = b0.x; B.u[1] = b0.y; B.u[2] = b0.z; B.u[3] = b0.w;
        B.u[4] = b1.x; B.u[5] = b1.y; B.u[6] = b1.z; B.u[7] = b1.w;
        acc[nt] = __builtin_amdgcn_wmma_f32_16x16x32_bf16(
            false, A.v, false, B.v, (short)0, acc[nt], false, false);
      }
    }
    if (lane < 16) {                           // D row M=0 lives in elem 0
#pragma unroll
      for (int nt = 0; nt < 6; ++nt) part[wave * G3 + nt * 16 + lane] = acc[nt][0];
    }
    __syncthreads();

    // ---- cross-wave split-K reduce + cheap input projection ----
    if (tid < G3) {
      float gh = bhs[tid];
#pragma unroll
      for (int w = 0; w < 8; ++w) gh += part[w * G3 + tid];
      float gx = bxs[tid];
#pragma unroll
      for (int d = 0; d < IN_DIMC; ++d) gx += xs[d] * Wxs[d * G3 + tid];
      gx_s[tid] = gx;
      gh_s[tid] = gh;
    }
    __syncthreads();

    // ---- gate nonlinearities + state update (32 units per WG) ----
    if (tid < HB) {
      float z    = 1.0f / (1.0f + __expf(-(gx_s[tid] + gh_s[tid])));
      float r    = 1.0f / (1.0f + __expf(-(gx_s[32 + tid] + gh_s[32 + tid])));
      float cand = tanhf(gx_s[64 + tid] + r * gh_s[64 + tid]);
      float hN   = z * h_loc[tid] + (1.0f - z) * cand;
      h_loc[tid] = hN;
      unsigned short b = f2bf(hN);
      hbits[tid] = b;
      hs[(size_t)t * HID + u0 + tid] = b;      // bf16 hs for final GEMM
    }
    __syncthreads();
    if (tid < HB / 2)
      hnxt[u0 / 2 + tid] =
          (unsigned)hbits[2 * tid] | ((unsigned)hbits[2 * tid + 1] << 16);

    // ---- grid barrier (monotone counter, sense-free) ----
    __syncthreads();
    if (tid == 0) {
      __threadfence();
      __hip_atomic_fetch_add(counter, 1u, __ATOMIC_ACQ_REL, __HIP_MEMORY_SCOPE_AGENT);
      const unsigned target = (unsigned)NWG * (unsigned)(t + 1);
      while (__hip_atomic_load(counter, __ATOMIC_ACQUIRE, __HIP_MEMORY_SCOPE_AGENT) < target)
        __builtin_amdgcn_s_sleep(1);
      __threadfence();
    }
    __syncthreads();
  }
}

// ---------------------------------------------------------------------------
// p = hs @ Wd + bd : dense WMMA bf16 GEMM. Block = 8 M-tiles x 1 N-tile;
// fragment-major B tile (8192 u32) staged once in LDS, read via ds_load_b128.
// ---------------------------------------------------------------------------
__global__ void __launch_bounds__(256, 2)
proj_kernel(const unsigned short* __restrict__ hs,  // [T,1024] bf16
            const unsigned* __restrict__ Wdp,       // [FRAGW] frag-major
            const float* __restrict__ bd,           // [96]
            float* __restrict__ out,                // [T,96]
            int T) {
  __shared__ unsigned Bt[KP * 16];   // 32 KB: this nt's fragments
  __shared__ float bds[16];

  const int tid  = threadIdx.x;
  const int lane = tid & 31;
  const int wave = tid >> 5;
  const int nt   = blockIdx.x % 6;
  const int mblk = blockIdx.x / 6;
  const int c0   = nt * 16;

  for (int i = tid; i < KP * 16; i += 256) {
    int kchunk = i >> 8, rem = i & 255;
    Bt[i] = Wdp[(kchunk * 6 + nt) * 256 + rem];
  }
  if (tid < 16) bds[tid] = bd[c0 + tid];
  __syncthreads();

  const int t0 = (mblk * 8 + wave) * 16;
  if (t0 >= T) return;

  const int half = lane >> 4;
  const int n    = lane & 15;
  const unsigned short* arow = hs + (size_t)(t0 + n) * HID + half * 16;

  v8f acc = {};
#pragma unroll 4
  for (int k0 = 0; k0 < HID; k0 += 32) {
    Frag A;
    uint4 lo = *reinterpret_cast<const uint4*>(arow + k0);        // K kb..kb+7
    uint4 hi = *reinterpret_cast<const uint4*>(arow + k0 + 8);    // K kb+8..kb+15
    A.u[0] = lo.x; A.u[1] = lo.y; A.u[2] = lo.z; A.u[3] = lo.w;
    A.u[4] = hi.x; A.u[5] = hi.y; A.u[6] = hi.z; A.u[7] = hi.w;
    const unsigned* fb = Bt + (k0 >> 5) * 256 + lane * 8;
    uint4 b0 = *(const uint4*)(fb);
    uint4 b1 = *(const uint4*)(fb + 4);
    Frag B;
    B.u[0] = b0.x; B.u[1] = b0.y; B.u[2] = b0.z; B.u[3] = b0.w;
    B.u[4] = b1.x; B.u[5] = b1.y; B.u[6] = b1.z; B.u[7] = b1.w;
    acc = __builtin_amdgcn_wmma_f32_16x16x32_bf16(
        false, A.v, false, B.v, (short)0, acc, false, false);
  }
#pragma unroll
  for (int r = 0; r < 8; ++r) {
    int row = t0 + r + half * 8;                 // D: VGPR r = row r / r+8
    out[(size_t)row * G3 + c0 + n] = acc[r] + bds[n];
  }
}

// ---------------------------------------------------------------------------
extern "C" void kernel_launch(void* const* d_in, const int* in_sizes, int n_in,
                              void* d_out, int out_size, void* d_ws, size_t ws_size,
                              hipStream_t stream) {
  const float* x  = (const float*)d_in[0];
  const float* Wx = (const float*)d_in[1];
  const float* Wh = (const float*)d_in[2];
  const float* bx = (const float*)d_in[3];
  const float* bh = (const float*)d_in[4];
  const float* Wd = (const float*)d_in[5];
  const float* bd = (const float*)d_in[6];
  float* out = (float*)d_out;
  const int T = in_sizes[0] / IN_DIMC;

  // workspace carve
  char* ws = (char*)d_ws;
  unsigned*       counter = (unsigned*)ws;                         // 128 B
  unsigned*       hbuf    = (unsigned*)(ws + 128);                 // 4 KB
  unsigned*       Wdp     = (unsigned*)(ws + 128 + 4096);          // 192 KB
  unsigned*       WhpG    = (unsigned*)(ws + 128 + 4096 + FRAGW * 4);
  unsigned short* hs      = (unsigned short*)
      (ws + 128 + 4096 + (size_t)FRAGW * 4 + (size_t)NWG * FRAGW * 4);

  init_ws_kernel<<<1024, 256, 0, stream>>>(Wh, Wd, counter, hbuf, Wdp, WhpG);

  // dynamic LDS: 51640 u32 = 206,560 B (< 320 KB WGP LDS)
  const size_t smem =
      (size_t)(FRAGW + KP + 8 * G3 + G3 + G3 + 8 * G3 + G3 + G3 +
               IN_DIMC + HB + HB / 2) * 4u;
  (void)hipFuncSetAttribute((const void*)gru_kernel,
                            hipFuncAttributeMaxDynamicSharedMemorySize,
                            (int)smem);
  gru_kernel<<<NWG, 256, smem, stream>>>(x, Wx, bx, bh, WhpG, counter, hbuf, hs, T);

  const int mblks = (T / 16 + 7) / 8;
  proj_kernel<<<mblks * 6, 256, 0, stream>>>(hs, Wdp, bd, out, T);
}
